// Generator_62629213110549
// MI455X (gfx1250) — compile-verified, matrix operand order checked by reading, not run on hardware
//
#include <hip/hip_runtime.h>
#include <hip/hip_bf16.h>

// ---------------------------------------------------------------------------
// Neural SDE (Euler-Maruyama scan of two 3-layer MLPs, HIDDEN=1024, B=2048)
// Heavy math on v_wmma_f32_16x16x32_f16; global->LDS staging via CDNA5
// async-load-to-LDS (ASYNCcnt) with a peeled 3-deep software pipeline.
// ---------------------------------------------------------------------------

typedef __attribute__((ext_vector_type(16))) _Float16 v16h;
typedef __attribute__((ext_vector_type(8)))  _Float16 v8h;
typedef __attribute__((ext_vector_type(8)))  float    v8f;
typedef __attribute__((ext_vector_type(4)))  int      v4i;

#define TILE_M 128
#define TILE_N 128
#define TILE_K 32
#define LDSS   40   // halfs per LDS row (32 data + 8 pad, keeps 16B alignment)

#if __has_builtin(__builtin_amdgcn_global_load_async_to_lds_b128)
#define SDE_ASYNC_LDS 1
#endif

typedef __attribute__((address_space(1))) v4i* sde_gv4i;
typedef __attribute__((address_space(3))) v4i* sde_lv4i;

__device__ __forceinline__ void sde_stage16(const _Float16* g, _Float16* l) {
#if defined(SDE_ASYNC_LDS)
    __builtin_amdgcn_global_load_async_to_lds_b128((sde_gv4i)g, (sde_lv4i)l, 0, 0);
#else
    *(v8h*)l = *(const v8h*)g;
#endif
}

template<int N>
__device__ __forceinline__ void sde_wait_async() {
#if __has_builtin(__builtin_amdgcn_s_wait_asynccnt)
    __builtin_amdgcn_s_wait_asynccnt(N);
#elif defined(SDE_ASYNC_LDS)
    asm volatile("s_wait_asynccnt %0" :: "i"(N) : "memory");
#endif
}

__device__ __forceinline__ float lipswish_f(float x) {
    // 0.909 * x * sigmoid(x)
    return 0.909f * x / (1.0f + __expf(-x));
}

// ---------------------------------------------------------------------------
// FULL-tile WMMA GEMM (no bounds checks; M%128==0, N%128==0, K%32==0, K>=64):
//   C[M,N] = act( A[M,K] * Bt[N,K]^T + bias[N] )
// Block = 256 threads = 8 wave32; block tile 128x128; wave tile 64x32.
// Peeled 3-deep LDS pipeline; global->LDS via async loads (ASYNCcnt).
// ---------------------------------------------------------------------------
template<bool ACT>
__global__ __launch_bounds__(256)
void sde_wmma_gemm_full(const _Float16* __restrict__ A, int lda,
                        const _Float16* __restrict__ Bt, int ldb,
                        const float* __restrict__ bias,
                        void* __restrict__ Cv, int ldc, int K)
{
    __shared__ _Float16 As[3][TILE_M * LDSS];
    __shared__ _Float16 Bs[3][TILE_N * LDSS];

    const int tid  = threadIdx.x;
    const int wave = tid >> 5;
    const int lane = tid & 31;
    const int lh   = lane >> 4;   // lane half (K-interleave select)
    const int lr   = lane & 15;

    const int wm = wave & 1;      // wave M position (2 x 64 rows)
    const int wn = wave >> 1;     // wave N position (4 x 32 cols)

    const int m0 = blockIdx.y * TILE_M;
    const int n0 = blockIdx.x * TILE_N;

    // staging geometry: 512 chunks of 16B per matrix tile, 2 per thread
    const int srow = tid >> 2;
    const int sc8  = (tid & 3) * 8;

    const _Float16* gA0 = A  + (size_t)(m0 + srow)      * lda + sc8;
    const _Float16* gA1 = A  + (size_t)(m0 + srow + 64) * lda + sc8;
    const _Float16* gB0 = Bt + (size_t)(n0 + srow)      * ldb + sc8;
    const _Float16* gB1 = Bt + (size_t)(n0 + srow + 64) * ldb + sc8;

    auto stage = [&](int buf, int k0) {
        sde_stage16(gA0 + k0, &As[buf][ srow       * LDSS + sc8]);
        sde_stage16(gA1 + k0, &As[buf][(srow + 64) * LDSS + sc8]);
        sde_stage16(gB0 + k0, &Bs[buf][ srow       * LDSS + sc8]);
        sde_stage16(gB1 + k0, &Bs[buf][(srow + 64) * LDSS + sc8]);
    };

    v8f acc[4][2];
#pragma unroll
    for (int mi = 0; mi < 4; ++mi)
#pragma unroll
        for (int ni = 0; ni < 2; ++ni)
            acc[mi][ni] = (v8f){0.f,0.f,0.f,0.f,0.f,0.f,0.f,0.f};

    // fragments per ISA 16-bit layout:
    // lane half 0: K 0..7 | 16..23 ; lane half 1: K 8..15 | 24..31
    auto compute = [&](const _Float16* as, const _Float16* bs) {
        v16h bfrag[2];
#pragma unroll
        for (int ni = 0; ni < 2; ++ni) {
            const int r = wn * 32 + ni * 16 + lr;
            v8h lo = *(const v8h*)(&bs[r * LDSS + lh * 8]);
            v8h hi = *(const v8h*)(&bs[r * LDSS + 16 + lh * 8]);
            bfrag[ni] = __builtin_shufflevector(lo, hi,
                0,1,2,3,4,5,6,7,8,9,10,11,12,13,14,15);
        }
#pragma unroll
        for (int mi = 0; mi < 4; ++mi) {
            const int r = wm * 64 + mi * 16 + lr;
            v8h lo = *(const v8h*)(&as[r * LDSS + lh * 8]);
            v8h hi = *(const v8h*)(&as[r * LDSS + 16 + lh * 8]);
            v16h afrag = __builtin_shufflevector(lo, hi,
                0,1,2,3,4,5,6,7,8,9,10,11,12,13,14,15);
#pragma unroll
            for (int ni = 0; ni < 2; ++ni) {
                acc[mi][ni] = __builtin_amdgcn_wmma_f32_16x16x32_f16(
                    false, afrag, false, bfrag[ni],
                    (short)0, acc[mi][ni], false, false);
            }
        }
    };

    const int ntiles = K / TILE_K;     // >= 2 at every call site
    stage(0, 0);
    stage(1, TILE_K);

    // steady state: always one tile in flight beyond the one being waited on
    int cmpBuf = 0, stgBuf = 2, kOff = 2 * TILE_K;
    for (int i = 0; i < ntiles - 2; ++i) {
        sde_wait_async<4>();           // tile i has landed (tile i+1 in flight)
        __syncthreads();
        stage(stgBuf, kOff);
        compute(As[cmpBuf], Bs[cmpBuf]);
        kOff += TILE_K;
        cmpBuf = (cmpBuf == 2) ? 0 : cmpBuf + 1;
        stgBuf = (stgBuf == 2) ? 0 : stgBuf + 1;
    }
    // tail iteration ntiles-2: tile ntiles-1 still in flight
    sde_wait_async<4>();
    __syncthreads();
    compute(As[cmpBuf], Bs[cmpBuf]);
    cmpBuf = (cmpBuf == 2) ? 0 : cmpBuf + 1;
    // tail iteration ntiles-1: nothing in flight
    sde_wait_async<0>();
    __syncthreads();
    compute(As[cmpBuf], Bs[cmpBuf]);

    // epilogue: bias (+LipSwish). C/D layout: VGPR r -> row r / r+8 by lane half
#pragma unroll
    for (int mi = 0; mi < 4; ++mi) {
#pragma unroll
        for (int ni = 0; ni < 2; ++ni) {
            const int col = n0 + wn * 32 + ni * 16 + lr;
            const float bv = bias[col];
#pragma unroll
            for (int r = 0; r < 8; ++r) {
                const int row = m0 + wm * 64 + mi * 16 + lh * 8 + r;
                float v = acc[mi][ni][r] + bv;
                if (ACT)
                    ((_Float16*)Cv)[(size_t)row * ldc + col] = (_Float16)lipswish_f(v);
                else
                    ((float*)Cv)[(size_t)row * ldc + col] = v;
            }
        }
    }
}

// ---------------------------------------------------------------------------
// Guarded WMMA GEMM (bounds-checked; used only for the N=64 decode GEMMs)
// ---------------------------------------------------------------------------
template<bool ACT>
__global__ __launch_bounds__(256)
void sde_wmma_gemm_gen(const _Float16* __restrict__ A, int lda,
                       const _Float16* __restrict__ Bt, int ldb,
                       const float* __restrict__ bias,
                       void* __restrict__ Cv, int ldc,
                       int M, int N, int K)
{
    __shared__ _Float16 As[TILE_M * LDSS];
    __shared__ _Float16 Bs[TILE_N * LDSS];

    const int tid  = threadIdx.x;
    const int wave = tid >> 5;
    const int lane = tid & 31;
    const int lh   = lane >> 4;
    const int lr   = lane & 15;
    const int wm = wave & 1;
    const int wn = wave >> 1;
    const int m0 = blockIdx.y * TILE_M;
    const int n0 = blockIdx.x * TILE_N;

    v8f acc[4][2];
#pragma unroll
    for (int mi = 0; mi < 4; ++mi)
#pragma unroll
        for (int ni = 0; ni < 2; ++ni)
            acc[mi][ni] = (v8f){0.f,0.f,0.f,0.f,0.f,0.f,0.f,0.f};

    for (int k0 = 0; k0 < K; k0 += TILE_K) {
#pragma unroll
        for (int it = 0; it < 2; ++it) {
            const int idx = tid + it * 256;
            const int row = idx >> 2;
            const int c8  = (idx & 3) * 8;
            v8h va = {};
            if (m0 + row < M) va = *(const v8h*)(A + (size_t)(m0 + row) * lda + k0 + c8);
            *(v8h*)(&As[row * LDSS + c8]) = va;
            v8h vb = {};
            if (n0 + row < N) vb = *(const v8h*)(Bt + (size_t)(n0 + row) * ldb + k0 + c8);
            *(v8h*)(&Bs[row * LDSS + c8]) = vb;
        }
        __syncthreads();

        v16h bfrag[2];
#pragma unroll
        for (int ni = 0; ni < 2; ++ni) {
            const int r = wn * 32 + ni * 16 + lr;
            v8h lo = *(const v8h*)(&Bs[r * LDSS + lh * 8]);
            v8h hi = *(const v8h*)(&Bs[r * LDSS + 16 + lh * 8]);
            bfrag[ni] = __builtin_shufflevector(lo, hi,
                0,1,2,3,4,5,6,7,8,9,10,11,12,13,14,15);
        }
#pragma unroll
        for (int mi = 0; mi < 4; ++mi) {
            const int r = wm * 64 + mi * 16 + lr;
            v8h lo = *(const v8h*)(&As[r * LDSS + lh * 8]);
            v8h hi = *(const v8h*)(&As[r * LDSS + 16 + lh * 8]);
            v16h afrag = __builtin_shufflevector(lo, hi,
                0,1,2,3,4,5,6,7,8,9,10,11,12,13,14,15);
#pragma unroll
            for (int ni = 0; ni < 2; ++ni) {
                acc[mi][ni] = __builtin_amdgcn_wmma_f32_16x16x32_f16(
                    false, afrag, false, bfrag[ni],
                    (short)0, acc[mi][ni], false, false);
            }
        }
        __syncthreads();
    }

#pragma unroll
    for (int mi = 0; mi < 4; ++mi) {
#pragma unroll
        for (int ni = 0; ni < 2; ++ni) {
            const int col = n0 + wn * 32 + ni * 16 + lr;
            if (col < N) {
                const float bv = bias[col];
#pragma unroll
                for (int r = 0; r < 8; ++r) {
                    const int row = m0 + wm * 64 + mi * 16 + lh * 8 + r;
                    if (row < M) {
                        float v = acc[mi][ni][r] + bv;
                        if (ACT)
                            ((_Float16*)Cv)[(size_t)row * ldc + col] = (_Float16)lipswish_f(v);
                        else
                            ((float*)Cv)[(size_t)row * ldc + col] = v;
                    }
                }
            }
        }
    }
}

// ---------------------------------------------------------------------------
// Small helper kernels
// ---------------------------------------------------------------------------
__global__ void sde_transpose_cvt(const float* __restrict__ W,
                                  _Float16* __restrict__ Wt, int K, int N)
{
    int idx = blockIdx.x * blockDim.x + threadIdx.x;
    if (idx >= K * N) return;
    int k = idx % K, n = idx / K;
    Wt[idx] = (_Float16)W[(size_t)k * N + n];
}

__global__ void sde_gather_cvt(const float* __restrict__ src,
                               _Float16* __restrict__ dst,
                               int rows, int cols, int srcStride)
{
    int idx = blockIdx.x * blockDim.x + threadIdx.x;
    if (idx >= rows * cols) return;
    int r = idx / cols, c = idx % cols;
    dst[idx] = (_Float16)src[(size_t)r * srcStride + c];
}

__global__ void sde_beff(const float* __restrict__ Wrow0,
                         const float* __restrict__ b,
                         const float* __restrict__ times,
                         float* __restrict__ beff, int steps, int n)
{
    int idx = blockIdx.x * blockDim.x + threadIdx.x;
    if (idx >= steps * n) return;
    int s = idx / n, j = idx % n;
    beff[idx] = b[j] + times[s] * Wrow0[j];
}

__global__ void sde_combine(float* __restrict__ z, _Float16* __restrict__ z16,
                            const float* __restrict__ drift,
                            const float* __restrict__ diff,
                            const float* __restrict__ noise,
                            const float* __restrict__ times, int s, int n)
{
    int i = blockIdx.x * blockDim.x + threadIdx.x;
    if (i >= n) return;
    float dt = times[s + 1] - times[s];
    float zn = z[i] + dt * (drift[i] + diff[i] * noise[i]);
    z[i]  = zn;
    z16[i] = (_Float16)zn;
}

// ---------------------------------------------------------------------------
extern "C" void kernel_launch(void* const* d_in, const int* in_sizes, int n_in,
                              void* d_out, int out_size, void* d_ws, size_t ws_size,
                              hipStream_t stream)
{
    (void)in_sizes; (void)n_in; (void)out_size; (void)ws_size;
    constexpr int Bsz = 2048, T = 21, IN = 64, H = 1024, OUT = 64, NSTEP = 20;

    const float* coeffs = (const float*)d_in[0];
    const float* times  = (const float*)d_in[1];
    const float* noise  = (const float*)d_in[2];
    const float* Wi     = (const float*)d_in[3];
    const float* bi     = (const float*)d_in[4];
    const float* Wd     = (const float*)d_in[5];
    const float* bd     = (const float*)d_in[6];
    const float* fWin   = (const float*)d_in[7];
    const float* fbin   = (const float*)d_in[8];
    const float* fWh    = (const float*)d_in[9];
    const float* fbh    = (const float*)d_in[10];
    const float* fWout  = (const float*)d_in[11];
    const float* fbout  = (const float*)d_in[12];
    const float* gWin   = (const float*)d_in[13];
    const float* gbin   = (const float*)d_in[14];
    const float* gWh    = (const float*)d_in[15];
    const float* gbh    = (const float*)d_in[16];
    const float* gWout  = (const float*)d_in[17];
    const float* gbout  = (const float*)d_in[18];

    float* out = (float*)d_out;

    char* ws = (char*)d_ws;
    size_t off = 0;
    auto carve = [&](size_t bytes) -> char* {
        char* p = ws + off;
        off += (bytes + 255) & ~(size_t)255;
        return p;
    };
    _Float16* wt[8];
    for (int i = 0; i < 8; ++i) wt[i] = (_Float16*)carve((size_t)H * H * 2);
    _Float16* wi_t  = (_Float16*)carve((size_t)H * IN * 2);
    _Float16* wd_t  = (_Float16*)carve((size_t)OUT * H * 2);
    _Float16* c0_16 = (_Float16*)carve((size_t)Bsz * IN * 2);
    float*    z32   = (float*)   carve((size_t)Bsz * H * 4);
    _Float16* z16   = (_Float16*)carve((size_t)Bsz * H * 2);
    _Float16* h16a  = (_Float16*)carve((size_t)Bsz * H * 2);
    _Float16* h16b  = (_Float16*)carve((size_t)Bsz * H * 2);
    float*    drift = (float*)   carve((size_t)Bsz * H * 4);
    float*    diffu = (float*)   carve((size_t)Bsz * H * 4);
    float*    beffF = (float*)   carve((size_t)NSTEP * H * 4);
    float*    beffG = (float*)   carve((size_t)NSTEP * H * 4);

    auto grid1 = [](int n) { return dim3((unsigned)((n + 255) / 256)); };

    // ---- one-time prep ---------------------------------------------------
    sde_transpose_cvt<<<grid1(H * H), 256, 0, stream>>>(fWin + H, wt[0], H, H);
    sde_transpose_cvt<<<grid1(H * H), 256, 0, stream>>>(fWh,                  wt[1], H, H);
    sde_transpose_cvt<<<grid1(H * H), 256, 0, stream>>>(fWh + (size_t)H * H,  wt[2], H, H);
    sde_transpose_cvt<<<grid1(H * H), 256, 0, stream>>>(fWout,                wt[3], H, H);
    sde_transpose_cvt<<<grid1(H * H), 256, 0, stream>>>(gWin + H, wt[4], H, H);
    sde_transpose_cvt<<<grid1(H * H), 256, 0, stream>>>(gWh,                  wt[5], H, H);
    sde_transpose_cvt<<<grid1(H * H), 256, 0, stream>>>(gWh + (size_t)H * H,  wt[6], H, H);
    sde_transpose_cvt<<<grid1(H * H), 256, 0, stream>>>(gWout,                wt[7], H, H);
    sde_transpose_cvt<<<grid1(IN * H), 256, 0, stream>>>(Wi, wi_t, IN, H);
    sde_transpose_cvt<<<grid1(H * OUT), 256, 0, stream>>>(Wd, wd_t, H, OUT);
    sde_beff<<<grid1(NSTEP * H), 256, 0, stream>>>(fWin, fbin, times, beffF, NSTEP, H);
    sde_beff<<<grid1(NSTEP * H), 256, 0, stream>>>(gWin, gbin, times, beffG, NSTEP, H);
    sde_gather_cvt<<<grid1(Bsz * IN), 256, 0, stream>>>(coeffs, c0_16, Bsz, IN, T * IN);

    const dim3 blk(256);
    const dim3 gFull(H / TILE_N, Bsz / TILE_M);                                 // 8 x 16
    const dim3 gDec((OUT + TILE_N - 1) / TILE_N, (Bsz + TILE_M - 1) / TILE_M);  // 1 x 16

    // ---- encode: z0 = coeffs0 @ Wi + bi ---------------------------------
    sde_wmma_gemm_full<false><<<gFull, blk, 0, stream>>>(c0_16, IN, wi_t, IN, bi,
                                                         z32, H, IN);
    sde_gather_cvt<<<grid1(Bsz * H), 256, 0, stream>>>(z32, z16, Bsz, H, H);

    // ---- decode t=0 ------------------------------------------------------
    sde_wmma_gemm_gen<false><<<gDec, blk, 0, stream>>>(z16, H, wd_t, H, bd,
                                                       out, T * OUT, Bsz, OUT, H);

    const float* fbh0 = fbh;  const float* fbh1 = fbh + H;
    const float* gbh0 = gbh;  const float* gbh1 = gbh + H;

    // ---- Euler-Maruyama scan --------------------------------------------
    for (int s = 0; s < NSTEP; ++s) {
        sde_wmma_gemm_full<true ><<<gFull, blk, 0, stream>>>(z16,  H, wt[0], H, beffF + (size_t)s * H, h16a, H, H);
        sde_wmma_gemm_full<true ><<<gFull, blk, 0, stream>>>(h16a, H, wt[1], H, fbh0,  h16b, H, H);
        sde_wmma_gemm_full<true ><<<gFull, blk, 0, stream>>>(h16b, H, wt[2], H, fbh1,  h16a, H, H);
        sde_wmma_gemm_full<false><<<gFull, blk, 0, stream>>>(h16a, H, wt[3], H, fbout, drift, H, H);
        sde_wmma_gemm_full<true ><<<gFull, blk, 0, stream>>>(z16,  H, wt[4], H, beffG + (size_t)s * H, h16a, H, H);
        sde_wmma_gemm_full<true ><<<gFull, blk, 0, stream>>>(h16a, H, wt[5], H, gbh0,  h16b, H, H);
        sde_wmma_gemm_full<true ><<<gFull, blk, 0, stream>>>(h16b, H, wt[6], H, gbh1,  h16a, H, H);
        sde_wmma_gemm_full<false><<<gFull, blk, 0, stream>>>(h16a, H, wt[7], H, gbout, diffu, H, H);
        sde_combine<<<grid1(Bsz * H), 256, 0, stream>>>(z32, z16, drift, diffu,
                                                        noise + (size_t)s * Bsz * H,
                                                        times, s, Bsz * H);
        sde_wmma_gemm_gen<false><<<gDec, blk, 0, stream>>>(z16, H, wd_t, H, bd,
                                                           out + (size_t)(s + 1) * OUT,
                                                           T * OUT, Bsz, OUT, H);
    }
}
